// Model_18691697672641
// MI455X (gfx1250) — matrix operand order repaired and checked by reading, not compiled, not run on hardware
//
#include <hip/hip_runtime.h>
#include <math.h>
#include <stdint.h>

// ---------------- model constants ----------------
#define BATCH   32
#define LSAMP   160000
#define NFFT    512
#define HOP     256
#define PADW    256
#define TOTAL   160512          // LSAMP + 2*PADW
#define TSTEPS  626
#define NF      257
#define TB      20032           // TSTEPS*BATCH
#define HID     512
#define G4      2048
#define KPAD_IN 288             // 257 -> mult of 32
#define NSPEC   576             // 514 -> mult of 64 (Re[0..256] | Im[0..256] | pad)
#define PROJN   5184            // 5140 -> mult of 64
#define PROJR   5140
#define BSROWS  40064           // 64 * TSTEPS
#define NWG_LSTM 32
#define LSTM_LDS (65536 + 8192) // 4x16x512 bf16 weights + 64x32 f32 gates

typedef __attribute__((ext_vector_type(16))) __bf16 v16bf;
typedef __attribute__((ext_vector_type(8)))  __bf16 v8bf;
typedef __attribute__((ext_vector_type(8)))  float  v8f;
typedef __attribute__((ext_vector_type(4)))  unsigned int u32x4;
typedef __attribute__((ext_vector_type(8)))  int i32x8;
typedef __attribute__((ext_vector_type(4)))  int i32x4;

// ---------------- workspace layout (bytes) ----------------
constexpr size_t ALN(size_t x) { return (x + 255) & ~size_t(255); }
constexpr size_t OFF_WSQN   = 0;
constexpr size_t OFF_DFTF   = ALN(OFF_WSQN   + (size_t)TOTAL * 4);
constexpr size_t OFF_IDFTB  = ALN(OFF_DFTF   + (size_t)NSPEC * NFFT * 2);
constexpr size_t OFF_FRAMES = ALN(OFF_IDFTB  + (size_t)NFFT * NSPEC * 2);
constexpr size_t OFF_SPEC   = ALN(OFF_FRAMES + (size_t)TB * NFFT * 2);
constexpr size_t OFF_HIN    = ALN(OFF_SPEC   + (size_t)TB * NSPEC * 4);
constexpr size_t OFF_WIH0T  = ALN(OFF_HIN    + (size_t)TB * KPAD_IN * 2);
constexpr size_t OFF_WHH0   = ALN(OFF_WIH0T  + (size_t)G4 * KPAD_IN * 2);
constexpr size_t OFF_WIH1T  = ALN(OFF_WHH0   + (size_t)G4 * HID * 2);
constexpr size_t OFF_WHH1   = ALN(OFF_WIH1T  + (size_t)G4 * HID * 2);
constexpr size_t OFF_PROJWT = ALN(OFF_WHH1   + (size_t)G4 * HID * 2);
constexpr size_t OFF_BIAS0  = ALN(OFF_PROJWT + (size_t)PROJN * HID * 2);
constexpr size_t OFF_BIAS1  = ALN(OFF_BIAS0  + (size_t)G4 * 4);
constexpr size_t OFF_PBIAS  = ALN(OFF_BIAS1  + (size_t)G4 * 4);
constexpr size_t OFF_XP     = ALN(OFF_PBIAS  + (size_t)PROJN * 4);
constexpr size_t OFF_HS0    = ALN(OFF_XP     + (size_t)TB * G4 * 4);
constexpr size_t OFF_HS1    = ALN(OFF_HS0    + (size_t)TB * HID * 2);
constexpr size_t OFF_COEF   = ALN(OFF_HS1    + (size_t)TB * HID * 2);
constexpr size_t OFF_IDFTA  = ALN(OFF_COEF   + (size_t)TB * PROJN * 2);
constexpr size_t OFF_FRAMO  = ALN(OFF_IDFTA  + (size_t)BSROWS * NSPEC * 2);
constexpr size_t OFF_ACC    = ALN(OFF_FRAMO  + (size_t)BSROWS * NFFT * 4);
constexpr size_t OFF_HCUR   = ALN(OFF_ACC    + (size_t)64 * TOTAL * 4);   // ping-pong 2x32x512 bf16
constexpr size_t OFF_BAR    = ALN(OFF_HCUR   + (size_t)2 * BATCH * HID * 2);

__device__ __forceinline__ float sigf(float x) { return 1.f / (1.f + expf(-x)); }
__device__ __forceinline__ float hannw(int n) {
  return 0.5f - 0.5f * cosf(6.28318530717958647692f * (float)n / 512.f);
}

// ---------------- tiny utility kernels ----------------
__global__ void zero_u32(unsigned* p, size_t n) {
  for (size_t i = blockIdx.x * 256ull + threadIdx.x; i < n; i += gridDim.x * 256ull) p[i] = 0u;
}

__global__ void wsqn_kernel(float* wsqn) {
  int l = blockIdx.x * 256 + threadIdx.x;
  if (l >= LSAMP) return;
  int p = l + PADW;
  int tlo = (p > 511) ? ((p - 511 + 255) >> 8) : 0;
  int thi = p >> 8; if (thi > TSTEPS - 1) thi = TSTEPS - 1;
  float s = 0.f;
  for (int t = tlo; t <= thi; ++t) {
    int n = p - (t << 8);
    if (n >= 0 && n < NFFT) { float w = hannw(n); s += w * w; }
  }
  wsqn[l] = (s > 1e-11f) ? s : 1.f;
}

// forward DFT basis, N-major (B^T): dftF[c][n], c = spectral col (Re|Im), n = time sample
__global__ void fwd_basis_kernel(__bf16* dftF) {
  int e = blockIdx.x * 256 + threadIdx.x;
  if (e >= NSPEC * NFFT) return;
  int c = e / NFFT, n = e % NFFT;
  float v = 0.f;
  if (c < NF)            v =  cosf(6.28318530717958647692f * (float)c * (float)n / 512.f);
  else if (c < 2 * NF) { int f = c - NF; v = -sinf(6.28318530717958647692f * (float)f * (float)n / 512.f); }
  dftF[e] = (__bf16)v;
}

// inverse DFT basis, N-major (B^T): idftB[n][r], n = time sample, r = spectral row (Re|Im)
__global__ void inv_basis_kernel(__bf16* idftB) {
  int e = blockIdx.x * 256 + threadIdx.x;
  if (e >= NFFT * NSPEC) return;
  int n = e / NSPEC, r = e % NSPEC;
  float v = 0.f;
  if (r < NF) {
    float sf = (r == 0 || r == 256) ? 1.f : 2.f;
    v = sf * cosf(6.28318530717958647692f * (float)r * (float)n / 512.f) / 512.f;
  } else if (r < 2 * NF) {
    int f = r - NF;
    float sf = (f == 0 || f == 256) ? 1.f : 2.f;
    v = -sf * sinf(6.28318530717958647692f * (float)f * (float)n / 512.f) / 512.f;
  }
  idftB[e] = (__bf16)v;
}

// pad-cast weights, keeping native (N x K) row-major = B^T layout for the GEMM
__global__ void wpad_kernel(const float* __restrict__ W, __bf16* __restrict__ out,
                            int Kreal, int Kp, int Nreal, int Np) {
  size_t e = blockIdx.x * 256ull + threadIdx.x;
  if (e >= (size_t)Np * Kp) return;
  int n = (int)(e / Kp), k = (int)(e % Kp);
  out[e] = (n < Nreal && k < Kreal) ? (__bf16)W[(size_t)n * Kreal + k] : (__bf16)0.f;
}

__global__ void wcopy_kernel(const float* __restrict__ W, __bf16* __restrict__ out, size_t n) {
  size_t e = blockIdx.x * 256ull + threadIdx.x;
  if (e < n) out[e] = (__bf16)W[e];
}

__global__ void bias_kernel(const float* __restrict__ a, const float* __restrict__ b,
                            float* __restrict__ out, int n, int nreal) {
  int i = blockIdx.x * 256 + threadIdx.x;
  if (i >= n) return;
  float v = 0.f;
  if (i < nreal) { v = a[i]; if (b) v += b[i]; }
  out[i] = v;
}

// ---------------- framing (reflect pad + window) ----------------
__global__ void frame_kernel(const float* __restrict__ x, __bf16* __restrict__ frames) {
  size_t e = blockIdx.x * 256ull + threadIdx.x;
  if (e >= (size_t)TB * NFFT) return;
  int n = (int)(e & 511);
  size_t r = e >> 9;                    // row = b*T + t
  int t = (int)(r % TSTEPS);
  int b = (int)(r / TSTEPS);
  int p = t * HOP + n - PADW;           // reflect-padded index
  if (p < 0) p = -p;
  if (p >= LSAMP) p = 2 * LSAMP - 2 - p;
  frames[e] = (__bf16)(x[(size_t)b * LSAMP + p] * hannw(n));
}

// ---------------- WMMA fragment helpers ----------------
__device__ __forceinline__ v16bf pack16(v8bf lo, v8bf hi) {
  v16bf r;
#pragma unroll
  for (int i = 0; i < 8; ++i) { r[i] = lo[i]; r[8 + i] = hi[i]; }
  return r;
}

// A fragment (16x32 bf16, row-major M x K): lane group g: 16 contiguous K at k0+16g
__device__ __forceinline__ v16bf load_a_frag(const __bf16* __restrict__ A, int lda, int m0, int k0) {
  int l = threadIdx.x & 31;
  const __bf16* p = A + (size_t)(m0 + (l & 15)) * lda + k0 + 16 * (l >> 4);
  return pack16(*(const v8bf*)p, *(const v8bf*)(p + 8));
}

// B fragment from N-major (B^T, N x K) global matrix: lane holds 16 contiguous K
__device__ __forceinline__ v16bf load_bt_frag(const __bf16* __restrict__ BT, int ldbt, int n0, int k0) {
  int l = threadIdx.x & 31;
  const __bf16* p = BT + (size_t)(n0 + (l & 15)) * ldbt + k0 + 16 * (l >> 4);
  return pack16(*(const v8bf*)p, *(const v8bf*)(p + 8));
}

// ---------------- generic bf16 GEMM: C = A(MxK) @ BT(NxK)^T + bias, optional tanh ----------------
// Block tile 64M x 64N, 8 waves, each wave: 16M x 32N (two f32 accumulators, shared A frag).
// B is consumed directly from global in transposed layout -> no LDS staging, no syncs.
__global__ void gemm_kernel(const __bf16* __restrict__ A, int lda,
                            const __bf16* __restrict__ BT, int ldbt,
                            float* __restrict__ Cf, __bf16* __restrict__ Cb, int ldc,
                            const float* __restrict__ bias,
                            int M, int N, int K, int Nreal, int act) {
  int w = threadIdx.x >> 5, l = threadIdx.x & 31;
  int mw = blockIdx.y * 64 + (w & 3) * 16;
  int nw = blockIdx.x * 64 + (w >> 2) * 32;
  v8f acc0 = {0.f, 0.f, 0.f, 0.f, 0.f, 0.f, 0.f, 0.f};
  v8f acc1 = {0.f, 0.f, 0.f, 0.f, 0.f, 0.f, 0.f, 0.f};

#pragma unroll 2
  for (int k0 = 0; k0 < K; k0 += 32) {
    v16bf a  = load_a_frag(A, lda, mw, k0);
    v16bf b0 = load_bt_frag(BT, ldbt, nw, k0);
    v16bf b1 = load_bt_frag(BT, ldbt, nw + 16, k0);
    acc0 = __builtin_amdgcn_wmma_f32_16x16x32_bf16(false, a, false, b0, (short)0, acc0, false, false);
    acc1 = __builtin_amdgcn_wmma_f32_16x16x32_bf16(false, a, false, b1, (short)0, acc1, false, false);
  }

#pragma unroll
  for (int r = 0; r < 8; ++r) {
    int m = mw + r + 8 * (l >> 4);
    if (m >= M) continue;
    int n0 = nw + (l & 15);
    int n1 = n0 + 16;
    if (n0 < Nreal) {
      float v = acc0[r] + (bias ? bias[n0] : 0.f);
      if (act) v = tanhf(v);
      if (Cb) Cb[(size_t)m * ldc + n0] = (__bf16)v;
      else    Cf[(size_t)m * ldc + n0] = v;
    }
    if (n1 < Nreal) {
      float v = acc1[r] + (bias ? bias[n1] : 0.f);
      if (act) v = tanhf(v);
      if (Cb) Cb[(size_t)m * ldc + n1] = (__bf16)v;
      else    Cf[(size_t)m * ldc + n1] = v;
    }
  }
}

// ---------------- magnitude^0.5 + LayerNorm over freq ----------------
__global__ void ln_kernel(const float* __restrict__ spec, const float* __restrict__ lng,
                          const float* __restrict__ lnb, __bf16* __restrict__ hin) {
  int warp = threadIdx.x >> 5, lane = threadIdx.x & 31;
  int q = blockIdx.x * 8 + warp;          // row = t*B + b
  if (q >= TB) return;
  int b = q & 31, t = q >> 5;
  const float* sr = spec + ((size_t)b * TSTEPS + t) * NSPEC;
  float s1 = 0.f, s2 = 0.f;
  for (int f = lane; f < NF; f += 32) {
    float re = sr[f], im = sr[NF + f];
    float a = sqrtf(sqrtf(re * re + im * im));   // |z|^0.5
    s1 += a; s2 += a * a;
  }
  for (int o = 16; o; o >>= 1) { s1 += __shfl_xor(s1, o, 32); s2 += __shfl_xor(s2, o, 32); }
  float mu = s1 / (float)NF;
  float var = s2 / (float)NF - mu * mu;
  float rs = rsqrtf(var + 1e-5f);
  __bf16* hr = hin + (size_t)q * KPAD_IN;
  for (int f = lane; f < KPAD_IN; f += 32) {
    float v = 0.f;
    if (f < NF) {
      float re = sr[f], im = sr[NF + f];
      float a = sqrtf(sqrtf(re * re + im * im));
      v = (a - mu) * rs * lng[f] + lnb[f];
    }
    hr[f] = (__bf16)v;
  }
}

// ---------------- grid-wide barrier ----------------
__device__ void grid_sync(unsigned* cnt, unsigned* gen, unsigned nwg) {
  __syncthreads();
  if (threadIdx.x == 0) {
    unsigned g = __hip_atomic_load(gen, __ATOMIC_ACQUIRE, __HIP_MEMORY_SCOPE_AGENT);
    unsigned prev = __hip_atomic_fetch_add(cnt, 1u, __ATOMIC_ACQ_REL, __HIP_MEMORY_SCOPE_AGENT);
    if (prev == nwg - 1) {
      __hip_atomic_store(cnt, 0u, __ATOMIC_RELAXED, __HIP_MEMORY_SCOPE_AGENT);
      __hip_atomic_store(gen, g + 1u, __ATOMIC_RELEASE, __HIP_MEMORY_SCOPE_AGENT);
    } else {
      while (__hip_atomic_load(gen, __ATOMIC_ACQUIRE, __HIP_MEMORY_SCOPE_AGENT) == g)
        __builtin_amdgcn_s_sleep(1);
    }
  }
  __syncthreads();
}

// ---------------- persistent LSTM recurrence ----------------
// Each of 32 WGs owns 16 hidden units; W_hh rows for (4 gates x 16 units) pinned in LDS via TDM.
// Computes G^T (64 local gate-rows x 32 batch) = Whh_rows(64x512) @ h^T(512x32) per step.
__global__ void lstm_kernel(const float* __restrict__ xp, const __bf16* __restrict__ Whh,
                            __bf16* __restrict__ hs, __bf16* __restrict__ hcur,
                            unsigned* cnt, unsigned* gen) {
  extern __shared__ __align__(16) char smem[];
  __bf16* Wl = (__bf16*)smem;                 // 4 stripes x (16 x 512) bf16 = 64KB
  float* gbuf = (float*)(smem + 65536);       // 64 x 32 f32 gates

  const int j = blockIdx.x;                   // hidden slice [j*16, j*16+16)
  const int tid = threadIdx.x;
  const int w = tid >> 5, l = tid & 31;

  // ---- preload weight stripes with the Tensor Data Mover: waves 0..3 issue one DMA each ----
  if (w < 4) {
    const int g = w;
    const __bf16* src = Whh + (size_t)(g * HID + j * 16) * HID;   // 16 contiguous rows of 512
    uint64_t ga = (uint64_t)(uintptr_t)src;
    u32x4 d0;
    d0[0] = 1u;                                   // count = 1 valid descriptor
    d0[1] = (unsigned)(g * 16384);                // lds_addr (bytes, dyn-LDS base 0)
    d0[2] = (unsigned)ga;                         // global_addr[31:0]
    d0[3] = (unsigned)((ga >> 32) & 0x1FFFFFFull) | (2u << 30);  // addr[56:32] | type=2
    i32x8 d1;
    d1[0] = (int)(1u << 16);          // data_size = 1 (2 bytes), wg_mask = 0
    d1[1] = (int)(16384u << 16);      // tensor_dim0 low16 (16384 elements)
    d1[2] = (int)(1u << 16);          // tensor_dim0 hi = 0 | tensor_dim1 = 1
    d1[3] = (int)(16384u << 16);      // tile_dim0 = 16384
    d1[4] = 1;                        // tile_dim1 = 1, tile_dim2 = 0
    d1[5] = 16384;                    // tensor_dim0_stride low32
    d1[6] = 0; d1[7] = 0;
    i32x4 dz; dz[0] = dz[1] = dz[2] = dz[3] = 0;
    i32x8 dz8; dz8[0]=dz8[1]=dz8[2]=dz8[3]=dz8[4]=dz8[5]=dz8[6]=dz8[7]=0;
    __builtin_amdgcn_tensor_load_to_lds(d0, d1, dz, dz, dz8, 0);
    __builtin_amdgcn_s_wait_tensorcnt(0);
  }
  __syncthreads();

  const int rt = w & 3;        // gate index for this wave
  const int nh = w >> 2;       // batch half (0: lanes see b 0..15, 1: b 16..31)
  float cst[2] = {0.f, 0.f};   // cell state, 2 cells per thread

  for (int t = 0; t < TSTEPS; ++t) {
    const __bf16* hprev = hcur + (size_t)(t & 1) * (BATCH * HID);
    __bf16* hnext = hcur + (size_t)((t + 1) & 1) * (BATCH * HID);
    const size_t xrow0 = (size_t)t * BATCH;

    // init accumulator from xp (input projection + biases)
    v8f acc;
#pragma unroll
    for (int r = 0; r < 8; ++r) {
      int m = r + 8 * (l >> 4);
      int gatecol = rt * HID + j * 16 + m;
      int b = nh * 16 + (l & 15);
      acc[r] = xp[(xrow0 + b) * G4 + gatecol];
    }
    if (t + 1 < TSTEPS) __builtin_prefetch(xp + (xrow0 + BATCH) * G4, 0, 1);

    // K loop: A = LDS weight stripe (16x512), B = h^T from global (bf16)
    const __bf16* Ws = Wl + (size_t)rt * 8192;
#pragma unroll 8
    for (int k = 0; k < HID; k += 32) {
      int kg = l >> 4;
      const __bf16* pa = Ws + (size_t)(l & 15) * HID + k + 16 * kg;
      v16bf af = pack16(*(const v8bf*)pa, *(const v8bf*)(pa + 8));
      const __bf16* pb = hprev + (size_t)(nh * 16 + (l & 15)) * HID + k + 16 * kg;
      v16bf bfv = pack16(*(const v8bf*)pb, *(const v8bf*)(pb + 8));
      acc = __builtin_amdgcn_wmma_f32_16x16x32_bf16(false, af, false, bfv, (short)0, acc, false, false);
    }

    // stage gates to LDS
#pragma unroll
    for (int r = 0; r < 8; ++r) {
      int m = r + 8 * (l >> 4);
      int b = nh * 16 + (l & 15);
      gbuf[(rt * 16 + m) * 32 + b] = acc[r];
    }
    __syncthreads();

    // cell update: 16 units x 32 batch = 512 cells, 2 per thread
#pragma unroll
    for (int i = 0; i < 2; ++i) {
      int e = tid * 2 + i;
      int b = e & 31, n = e >> 5;
      float ig = gbuf[(n) * 32 + b];
      float fg = gbuf[(16 + n) * 32 + b];
      float gg = gbuf[(32 + n) * 32 + b];
      float og = gbuf[(48 + n) * 32 + b];
      float c = sigf(fg) * cst[i] + sigf(ig) * tanhf(gg);
      float h = sigf(og) * tanhf(c);
      cst[i] = c;
      int hidx = j * 16 + n;
      hnext[(size_t)b * HID + hidx] = (__bf16)h;
      hs[(xrow0 + b) * HID + hidx] = (__bf16)h;
    }
    __threadfence();
    grid_sync(cnt, gen, NWG_LSTM);
    __builtin_amdgcn_s_cluster_barrier();   // NOP outside a cluster; device-wide already synced
  }
}

// ---------------- complex FIR filter -> iDFT input matrix ----------------
__global__ void filter_kernel(const float* __restrict__ spec, const __bf16* __restrict__ coef,
                              __bf16* __restrict__ idftA) {
  size_t e = blockIdx.x * 256ull + threadIdx.x;
  if (e >= (size_t)64 * NF * TSTEPS) return;
  int bs = (int)(e / (NF * TSTEPS));
  int rem = (int)(e % (NF * TSTEPS));
  int f = rem / TSTEPS, t = rem % TSTEPS;
  int b = bs >> 1, s = bs & 1;
  float er = 0.f, ei = 0.f;
  const size_t crow = ((size_t)t * BATCH + b) * PROJN;
#pragma unroll
  for (int d = 0; d < 5; ++d) {
    int td = t - 4 + d;
    if (td < 0) continue;
    size_t sr = ((size_t)b * TSTEPS + td) * NSPEC;
    float xr = spec[sr + f], xi = spec[sr + NF + f];
    float cr = (float)coef[crow + (size_t)((d * 2 + s) * NF + f)];
    float ci = (float)coef[crow + (size_t)(((5 + d) * 2 + s) * NF + f)];
    er += xr * cr - xi * ci;
    ei += xr * ci + xi * cr;
  }
  size_t ro = ((size_t)bs * TSTEPS + t) * NSPEC;
  idftA[ro + f] = (__bf16)er;
  idftA[ro + NF + f] = (__bf16)ei;
}

// ---------------- overlap-add ----------------
__global__ void ola_kernel(const float* __restrict__ framesO, float* __restrict__ acc) {
  size_t e = blockIdx.x * 256ull + threadIdx.x;
  if (e >= (size_t)BSROWS * NFFT) return;
  int n = (int)(e & 511);
  size_t ro = e >> 9;
  int t = (int)(ro % TSTEPS);
  int bs = (int)(ro / TSTEPS);
  atomicAdd(&acc[(size_t)bs * TOTAL + t * HOP + n], framesO[e] * hannw(n));
}

__global__ void final_kernel(const float* __restrict__ acc, const float* __restrict__ wsqn,
                             float* __restrict__ out) {
  size_t e = blockIdx.x * 256ull + threadIdx.x;
  if (e >= (size_t)64 * LSAMP) return;
  int bs = (int)(e / LSAMP);
  int l0 = (int)(e % LSAMP);
  out[e] = acc[(size_t)bs * TOTAL + PADW + l0] / wsqn[l0];
}

// ---------------- host driver ----------------
static inline unsigned cdiv(size_t a, size_t b) { return (unsigned)((a + b - 1) / b); }

extern "C" void kernel_launch(void* const* d_in, const int* in_sizes, int n_in,
                              void* d_out, int out_size, void* d_ws, size_t ws_size,
                              hipStream_t stream) {
  const float* x     = (const float*)d_in[0];
  const float* ln_g  = (const float*)d_in[1];
  const float* ln_b  = (const float*)d_in[2];
  const float* W_ih0 = (const float*)d_in[3];
  const float* W_hh0 = (const float*)d_in[4];
  const float* b_ih0 = (const float*)d_in[5];
  const float* b_hh0 = (const float*)d_in[6];
  const float* W_ih1 = (const float*)d_in[7];
  const float* W_hh1 = (const float*)d_in[8];
  const float* b_ih1 = (const float*)d_in[9];
  const float* b_hh1 = (const float*)d_in[10];
  const float* projW = (const float*)d_in[11];
  const float* projB = (const float*)d_in[12];
  float* out = (float*)d_out;
  char* ws = (char*)d_ws;

  float*  wsqn   = (float*)(ws + OFF_WSQN);
  __bf16* dftF   = (__bf16*)(ws + OFF_DFTF);
  __bf16* idftB  = (__bf16*)(ws + OFF_IDFTB);
  __bf16* frames = (__bf16*)(ws + OFF_FRAMES);
  float*  spec   = (float*)(ws + OFF_SPEC);
  __bf16* hin    = (__bf16*)(ws + OFF_HIN);
  __bf16* wih0t  = (__bf16*)(ws + OFF_WIH0T);
  __bf16* whh0   = (__bf16*)(ws + OFF_WHH0);
  __bf16* wih1t  = (__bf16*)(ws + OFF_WIH1T);
  __bf16* whh1   = (__bf16*)(ws + OFF_WHH1);
  __bf16* projwt = (__bf16*)(ws + OFF_PROJWT);
  float*  bias0  = (float*)(ws + OFF_BIAS0);
  float*  bias1  = (float*)(ws + OFF_BIAS1);
  float*  pbias  = (float*)(ws + OFF_PBIAS);
  float*  xp     = (float*)(ws + OFF_XP);
  __bf16* hs0    = (__bf16*)(ws + OFF_HS0);
  __bf16* hs1    = (__bf16*)(ws + OFF_HS1);
  __bf16* coef   = (__bf16*)(ws + OFF_COEF);
  __bf16* idftA  = (__bf16*)(ws + OFF_IDFTA);
  float*  framO  = (float*)(ws + OFF_FRAMO);
  float*  accb   = (float*)(ws + OFF_ACC);
  __bf16* hcur   = (__bf16*)(ws + OFF_HCUR);
  unsigned* bar  = (unsigned*)(ws + OFF_BAR);

  (void)hipFuncSetAttribute((const void*)lstm_kernel, hipFuncAttributeMaxDynamicSharedMemorySize, LSTM_LDS);

  // ---- prep ----
  wsqn_kernel<<<cdiv(LSAMP, 256), 256, 0, stream>>>(wsqn);
  fwd_basis_kernel<<<cdiv((size_t)NSPEC * NFFT, 256), 256, 0, stream>>>(dftF);
  inv_basis_kernel<<<cdiv((size_t)NFFT * NSPEC, 256), 256, 0, stream>>>(idftB);
  // weights stay in native (N x K) = B^T layout; just pad K and cast to bf16
  wpad_kernel<<<cdiv((size_t)G4 * KPAD_IN, 256), 256, 0, stream>>>(W_ih0, wih0t, NF, KPAD_IN, G4, G4);
  wpad_kernel<<<cdiv((size_t)G4 * HID, 256), 256, 0, stream>>>(W_ih1, wih1t, HID, HID, G4, G4);
  wpad_kernel<<<cdiv((size_t)PROJN * HID, 256), 256, 0, stream>>>(projW, projwt, HID, HID, PROJR, PROJN);
  wcopy_kernel<<<cdiv((size_t)G4 * HID, 256), 256, 0, stream>>>(W_hh0, whh0, (size_t)G4 * HID);
  wcopy_kernel<<<cdiv((size_t)G4 * HID, 256), 256, 0, stream>>>(W_hh1, whh1, (size_t)G4 * HID);
  bias_kernel<<<cdiv(G4, 256), 256, 0, stream>>>(b_ih0, b_hh0, bias0, G4, G4);
  bias_kernel<<<cdiv(G4, 256), 256, 0, stream>>>(b_ih1, b_hh1, bias1, G4, G4);
  bias_kernel<<<cdiv(PROJN, 256), 256, 0, stream>>>(projB, nullptr, pbias, PROJN, PROJR);

  // ---- STFT as GEMM ----
  frame_kernel<<<cdiv((size_t)TB * NFFT, 256), 256, 0, stream>>>(x, frames);
  gemm_kernel<<<dim3(NSPEC / 64, TB / 64), 256, 0, stream>>>(
      frames, NFFT, dftF, NFFT, spec, nullptr, NSPEC, nullptr, TB, NSPEC, NFFT, 2 * NF, 0);

  // ---- layernorm ----
  ln_kernel<<<TB / 8, 256, 0, stream>>>(spec, ln_g, ln_b, hin);

  // ---- LSTM layer 0 ----
  gemm_kernel<<<dim3(G4 / 64, TB / 64), 256, 0, stream>>>(
      hin, KPAD_IN, wih0t, KPAD_IN, xp, nullptr, G4, bias0, TB, G4, KPAD_IN, G4, 0);
  zero_u32<<<64, 256, 0, stream>>>((unsigned*)hcur, (2 * BATCH * HID * 2) / 4 + 64);
  lstm_kernel<<<NWG_LSTM, 256, LSTM_LDS, stream>>>(xp, whh0, hs0, hcur, bar, bar + 1);

  // ---- LSTM layer 1 ----
  gemm_kernel<<<dim3(G4 / 64, TB / 64), 256, 0, stream>>>(
      hs0, HID, wih1t, HID, xp, nullptr, G4, bias1, TB, G4, HID, G4, 0);
  zero_u32<<<64, 256, 0, stream>>>((unsigned*)hcur, (2 * BATCH * HID * 2) / 4 + 64);
  lstm_kernel<<<NWG_LSTM, 256, LSTM_LDS, stream>>>(xp, whh1, hs1, hcur, bar, bar + 1);

  // ---- projection + tanh ----
  gemm_kernel<<<dim3(PROJN / 64, TB / 64), 256, 0, stream>>>(
      hs1, HID, projwt, HID, nullptr, coef, PROJN, pbias, TB, PROJN, HID, PROJR, 1);

  // ---- complex FIR filter ----
  zero_u32<<<2048, 256, 0, stream>>>((unsigned*)idftA, (size_t)BSROWS * NSPEC * 2 / 4);
  filter_kernel<<<cdiv((size_t)64 * NF * TSTEPS, 256), 256, 0, stream>>>(spec, coef, idftA);

  // ---- iSTFT as GEMM + overlap-add ----
  gemm_kernel<<<dim3(NFFT / 64, BSROWS / 64), 256, 0, stream>>>(
      idftA, NSPEC, idftB, NSPEC, framO, nullptr, NFFT, nullptr, BSROWS, NFFT, NSPEC, NFFT, 0);
  zero_u32<<<2048, 256, 0, stream>>>((unsigned*)accb, (size_t)64 * TOTAL);
  ola_kernel<<<cdiv((size_t)BSROWS * NFFT, 256), 256, 0, stream>>>(framO, accb);
  final_kernel<<<cdiv((size_t)64 * LSAMP, 256), 256, 0, stream>>>(accb, wsqn, out);
}
